// PSFM_5102421147994
// MI455X (gfx1250) — matrix operand order repaired and checked by reading, not compiled, not run on hardware
//
#include <hip/hip_runtime.h>

// ============================================================================
// PSFM (dense blocks + GEFM cross-attention) for MI455X / gfx1250.
// f32 -> f16 NHWC once; convs = implicit GEMM on v_wmma_f32_16x16x32_f16 with
// NBLK-wide N register blocking; the two 4096x4096 attentions are flash-style
// (online softmax, mask never materialized).  All loads are branch-free
// (clamped addresses + v_cndmask zeroing) and 16-byte vectorized; attention
// Q tiles are staged into LDS by the Tensor Data Mover (tensor_load_to_lds +
// s_wait_tensorcnt) when available, V tiles cooperatively (transposed).
// ============================================================================

typedef __attribute__((ext_vector_type(16))) _Float16     v16h;
typedef __attribute__((ext_vector_type(8)))  float        v8f;
typedef __attribute__((ext_vector_type(4)))  int          v4i;
typedef __attribute__((ext_vector_type(4)))  unsigned int v4u;
typedef __attribute__((ext_vector_type(8)))  int          v8i;

union AF { v16h v; _Float16 h[16]; unsigned int u[8]; v4i q[2]; };

#define BATCH 4
#define HSZ 64
#define WSZ 64
#define PIX_PER_IMG 4096
#define NPIX 16384          // BATCH * H * W
#define ATT_C 128
#define BN_EPSF 1e-5f

#if __has_builtin(__builtin_amdgcn_tensor_load_to_lds)
#define HAVE_TDM 1
#else
#define HAVE_TDM 0
#endif

// 16-bit A-matrix 16x32 layout (CDNA5 ISA 7.12.2): u[0..3] covers
// K [8h, 8h+8), u[4..7] covers K [16+8h, 16+8h+8): two 16-byte runs.
__device__ __forceinline__ v8f wmma16(const AF& a, const AF& b, v8f c) {
  return __builtin_amdgcn_wmma_f32_16x16x32_f16(
      /*neg_a=*/false, a.v, /*neg_b=*/false, b.v,
      /*c_mod=*/(short)0, c, /*reuse_a=*/false, /*reuse_b=*/false);
}

// ---------------------------------------------------------------------------
// NCHW f32 -> NHWC f16 (C = 256 fixed)
// ---------------------------------------------------------------------------
__global__ __launch_bounds__(256) void to_nhwc256(const float* __restrict__ in,
                                                  _Float16* __restrict__ out) {
  int i = blockIdx.x * blockDim.x + threadIdx.x;
  const int tot = BATCH * 256 * PIX_PER_IMG;
  if (i >= tot) return;
  int b = i >> 20;
  int r = i & ((1 << 20) - 1);
  int c = r >> 12;
  int pix = r & 4095;
  out[((size_t)(b << 12) + pix) * 256 + c] = (_Float16)in[i];
}

// ---------------------------------------------------------------------------
// OIHW f32 weights -> [O][K] f16, K = (ky*KS+kx)*Ci + c  (matches im2col order)
// ---------------------------------------------------------------------------
__global__ __launch_bounds__(256) void pack_weights(const float* __restrict__ w,
                                                    _Float16* __restrict__ out,
                                                    int O, int Ci, int KS) {
  int i = blockIdx.x * blockDim.x + threadIdx.x;
  int per_o = Ci * KS * KS;
  int tot = O * per_o;
  if (i >= tot) return;
  int o = i / per_o;
  int r = i - o * per_o;
  int kk = r / Ci;
  int c = r - kk * Ci;
  int ky = kk / KS;
  int kx = kk - ky * KS;
  out[i] = (_Float16)w[((o * Ci + c) * KS + ky) * KS + kx];
}

// ---------------------------------------------------------------------------
// Implicit-GEMM conv: 16(M) x NBLK*16(N) per wave, K loop step 32 (f16 WMMA).
// Virtual concat of two NHWC input regions (A then B along channels).
// NGSHIFT = log2(#n-groups): tile decode is pure shifts/masks.
// mode 0: +bias (p0).  mode 1: BN (p0=scale,p1=bias,p2=mean,p3=var) + ReLU.
// outF32 != null: store f32 NCHW (final output), else f16 NHWC strided.
// ---------------------------------------------------------------------------
template <int NBLK, int NGSHIFT>
__global__ __launch_bounds__(256) void conv_gemm(
    const _Float16* __restrict__ inA, int cA, int strideA, int offA,
    const _Float16* __restrict__ inB, int cB, int strideB, int offB,
    const _Float16* __restrict__ wp, int KS, int Cout,
    const float* __restrict__ p0, const float* __restrict__ p1,
    const float* __restrict__ p2, const float* __restrict__ p3,
    int mode,
    _Float16* __restrict__ outH, int outStride, int outOff,
    float* __restrict__ outF32)
{
  const int wid  = (blockIdx.x * blockDim.x + threadIdx.x) >> 5;
  const int lane = threadIdx.x & 31;
  const int half = lane >> 4;
  const int l15  = lane & 15;
  const int tileM = wid >> NGSHIFT;
  const int tileG = wid & ((1 << NGSHIFT) - 1);
  const int m0 = tileM << 4;
  const int n0 = tileG * (NBLK << 4);
  if (m0 >= NPIX) return;

  const int Ctot = cA + cB;                 // always a multiple of 32 here
  const int Ktot = KS * KS * Ctot;
  const int pad  = KS >> 1;

  const int mA = m0 + l15;                  // this lane's A row (pixel)
  const int bA = mA >> 12;
  const int yA = (mA >> 6) & 63;
  const int xA = mA & 63;

  v8f acc[NBLK];
#pragma unroll
  for (int t = 0; t < NBLK; ++t) acc[t] = (v8f){0.f,0.f,0.f,0.f,0.f,0.f,0.f,0.f};

  int ky = 0, kx = 0, cbase = 0;            // incremental K decode
  for (int k0 = 0; k0 < Ktot; k0 += 32) {
    const int py = yA + ky - pad;
    const int px = xA + kx - pad;
    const bool inb = ((unsigned)py < (unsigned)HSZ) && ((unsigned)px < (unsigned)WSZ);
    const int pyc = py < 0 ? 0 : (py > 63 ? 63 : py);   // clamp: always valid addr
    const int pxc = px < 0 ? 0 : (px > 63 ? 63 : px);
    const int pix = (bA << 12) + (pyc << 6) + pxc;
    const _Float16* src;
    int base;
    if (cbase < cA) { src = inA; base = pix * strideA + offA + cbase; }
    else            { src = inB; base = pix * strideB + offB + (cbase - cA); }

    AF a;
    a.q[0] = *(const v4i*)(src + base + (half << 3));        // K [8h, 8h+8)
    a.q[1] = *(const v4i*)(src + base + 16 + (half << 3));   // K [16+8h, ..+8)
#pragma unroll
    for (int p = 0; p < 8; ++p) a.u[p] = inb ? a.u[p] : 0u;  // v_cndmask, no branch

#pragma unroll
    for (int t = 0; t < NBLK; ++t) {
      // B-matrix (32x16, 16-bit): same K order as A: K [8h, 8h+8) then [16+8h, ..+8).
      const int brow = (n0 + (t << 4) + l15) * Ktot + k0 + (half << 3);
      AF bfr;
      bfr.q[0] = *(const v4i*)(wp + brow);
      bfr.q[1] = *(const v4i*)(wp + brow + 16);
      acc[t] = wmma16(a, bfr, acc[t]);
    }

    cbase += 32;
    if (cbase >= Ctot) { cbase = 0; ++kx; if (kx >= KS) { kx = 0; ++ky; } }
  }

  // Epilogue.  C/D layout: lane holds one column, rows v + 8*half.
#pragma unroll
  for (int t = 0; t < NBLK; ++t) {
    const int n = n0 + (t << 4) + l15;
    float mul = 1.f, add;
    if (mode == 1) {
      const float inv = rsqrtf(p3[n] + BN_EPSF) * p0[n];
      mul = inv; add = p1[n] - p2[n] * inv;
    } else {
      add = p0[n];
    }
#pragma unroll
    for (int v = 0; v < 8; ++v) {
      const int mr = m0 + v + (half << 3);
      float val = acc[t][v] * mul + add;
      if (mode == 1) val = fmaxf(val, 0.f);
      if (outF32) {
        const int bb = mr >> 12, rem = mr & 4095;
        outF32[((size_t)(bb * Cout + n) << 12) + rem] = val;   // NCHW f32
      } else {
        outH[(size_t)mr * outStride + outOff + n] = (_Float16)val;
      }
    }
  }
}

// ---------------------------------------------------------------------------
// Flash attention: mask = softmax(K^T Q) ; out = gamma * (mask V^T) + other.
// One wave owns a (batch, 16-row) query tile; all 8 waves of a block share the
// same batch + j sweep: Q[32][128] tile staged by the TDM (one descriptor per
// j-step, wave 0), V^T[128][32] staged cooperatively.  K^T A-fragments are
// j-invariant and hoisted into registers.  NHWC f16 [B][4096][128] tensors.
// ---------------------------------------------------------------------------
__global__ __launch_bounds__(256) void attend_flash(
    const _Float16* __restrict__ kf, const _Float16* __restrict__ qf,
    const _Float16* __restrict__ vf, const _Float16* __restrict__ other,
    const float* __restrict__ gamma, _Float16* __restrict__ out)
{
  __shared__ __align__(16) _Float16 Qs[32][ATT_C];     // 8 KB  [j][c]
  __shared__ __align__(16) _Float16 VTs[ATT_C][32];    // 8 KB  [c][j]
  __shared__ __align__(16) _Float16 Ps[8][16 * 32];    // 8 KB  per-wave P
  const int wslot = threadIdx.x >> 5;
  const int wid = blockIdx.x * 8 + wslot;              // 1024 waves total
  const int lane = threadIdx.x & 31;
  const int half = lane >> 4;
  const int l15 = lane & 15;
  const int b = wid >> 8;                              // 256 row-tiles / batch
  const int i0 = (wid & 255) << 4;
  const int C = ATT_C, N = PIX_PER_IMG;

  const _Float16* kb = kf + (size_t)b * N * C;
  const _Float16* qb = qf + (size_t)b * N * C;
  const _Float16* vb = vf + (size_t)b * N * C;

  // Hoisted A-fragments: K^T rows i0..i0+15, 4 c-steps of 32.
  AF aK[4];
#pragma unroll
  for (int ci = 0; ci < 4; ++ci) {
    const _Float16* kp = kb + (size_t)(i0 + l15) * C + (ci << 5);
    aK[ci].q[0] = *(const v4i*)(kp + (half << 3));
    aK[ci].q[1] = *(const v4i*)(kp + 16 + (half << 3));
  }

  v8f O[8];
#pragma unroll
  for (int t = 0; t < 8; ++t) O[t] = (v8f){0.f,0.f,0.f,0.f,0.f,0.f,0.f,0.f};
  float mrow[8], lrow[8];
#pragma unroll
  for (int v = 0; v < 8; ++v) { mrow[v] = -1e30f; lrow[v] = 0.f; }

  for (int j0 = 0; j0 < N; j0 += 32) {
#if HAVE_TDM
    // ---- Q tile via Tensor Data Mover: flat 32x128 f16 (8 KB) copy ----
    if (wslot == 0) {
      const unsigned long long ga =
          (unsigned long long)(const void*)(qb + (size_t)j0 * C);
      const unsigned ldsOff =
          (unsigned)(unsigned long long)(const void*)(&Qs[0][0]);
      v4u g0;                                   // D# group 0 (ISA 8.3)
      g0.x = 1u;                                // count=1, user descriptor
      g0.y = ldsOff;                            // lds_addr (bytes)
      g0.z = (unsigned)ga;                      // global_addr[31:0]
      g0.w = (unsigned)((ga >> 32) & 0x01FFFFFFu) | 0x80000000u; // [56:32]|type=2
      v8i g1;                                   // D# group 1 (ISA 8.4)
      g1[0] = 0x10000;                          // data_size=2B; no multicast
      g1[1] = (int)(4096u << 16);               // tensor_dim0[15:0]=4096
      g1[2] = (int)(1u << 16);                  // dim0[31:16]=0, tensor_dim1=1
      g1[3] = (int)(4096u << 16);               // tile_dim0=4096 elements
      g1[4] = 1;                                // tile_dim1=1
      g1[5] = 4096;                             // tensor_dim0_stride
      g1[6] = 0; g1[7] = 0;
      v4i g2 = {0, 0, 0, 0};                    // groups 2/3 unused (2-D tile)
      v4i g3 = {0, 0, 0, 0};
      v8i g4 = {0, 0, 0, 0, 0, 0, 0, 0};        // trailing group (clang-23 form)
      __builtin_amdgcn_tensor_load_to_lds(g0, g1, g2, g3, g4, 0);
    }
#endif
    // ---- cooperative stage of V^T[128][32] (and Q fallback) ----
#pragma unroll
    for (int it = 0; it < 2; ++it) {
      const int idx = threadIdx.x + (it << 8);         // 0..511
      const int j = idx >> 4;                          // 0..31
      const int c0 = (idx & 15) << 3;                  // 0..120 step 8
#if !HAVE_TDM
      *(v4i*)(&Qs[j][c0]) = *(const v4i*)(qb + (size_t)(j0 + j) * C + c0);
#endif
      AF t16;
      t16.q[0] = *(const v4i*)(vb + (size_t)(j0 + j) * C + c0);
#pragma unroll
      for (int e = 0; e < 8; ++e) VTs[c0 + e][j] = t16.h[e];
      if (j0 + 32 < N) {                               // prefetch next tile
        __builtin_prefetch(qb + (size_t)(j0 + 32 + j) * C + c0, 0, 0);
        __builtin_prefetch(vb + (size_t)(j0 + 32 + j) * C + c0, 0, 0);
      }
    }
#if HAVE_TDM
    if (wslot == 0) __builtin_amdgcn_s_wait_tensorcnt(0);
#endif
    __syncthreads();

    // ---- scores S[16 x 32] = K^T(i,c) * Q(c,j): B-fragments from LDS ----
    v8f S0 = (v8f){0.f,0.f,0.f,0.f,0.f,0.f,0.f,0.f};
    v8f S1 = S0;
#pragma unroll
    for (int ci = 0; ci < 4; ++ci) {
      const int c0 = ci << 5;
      AF bq;
      bq.q[0] = *(const v4i*)(&Qs[l15][c0 + (half << 3)]);
      bq.q[1] = *(const v4i*)(&Qs[l15][c0 + 16 + (half << 3)]);
      S0 = wmma16(aK[ci], bq, S0);
      bq.q[0] = *(const v4i*)(&Qs[16 + l15][c0 + (half << 3)]);
      bq.q[1] = *(const v4i*)(&Qs[16 + l15][c0 + 16 + (half << 3)]);
      S1 = wmma16(aK[ci], bq, S1);
    }

    // ---- online softmax over j (rows live per half-wave) ----
    float scale[8];
#pragma unroll
    for (int v = 0; v < 8; ++v) {
      float mx = fmaxf(S0[v], S1[v]);
#pragma unroll
      for (int d = 1; d < 16; d <<= 1) mx = fmaxf(mx, __shfl_xor(mx, d, 32));
      const float mnew = fmaxf(mrow[v], mx);
      scale[v] = __expf(mrow[v] - mnew);
      const float e0 = __expf(S0[v] - mnew);
      const float e1 = __expf(S1[v] - mnew);
      float rs = e0 + e1;
#pragma unroll
      for (int d = 1; d < 16; d <<= 1) rs += __shfl_xor(rs, d, 32);
      lrow[v] = lrow[v] * scale[v] + rs;
      mrow[v] = mnew;
      const int row = v + (half << 3);
      Ps[wslot][row * 32 + l15]      = (_Float16)e0;
      Ps[wslot][row * 32 + 16 + l15] = (_Float16)e1;
    }
#pragma unroll
    for (int ct = 0; ct < 8; ++ct)
#pragma unroll
      for (int v = 0; v < 8; ++v) O[ct][v] *= scale[v];

    __syncthreads();   // uniform; orders P writes before A-layout reads

    AF pa;
    pa.q[0] = *(const v4i*)(&Ps[wslot][l15 * 32 + (half << 3)]);
    pa.q[1] = *(const v4i*)(&Ps[wslot][l15 * 32 + 16 + (half << 3)]);

    // ---- O[16 x 128] += P(16 x 32 j) * V^T(32 j x 16 c), B from LDS ----
#pragma unroll
    for (int ct = 0; ct < 8; ++ct) {
      const int cc = (ct << 4) + l15;
      AF vbf;
      vbf.q[0] = *(const v4i*)(&VTs[cc][(half << 3)]);
      vbf.q[1] = *(const v4i*)(&VTs[cc][16 + (half << 3)]);
      O[ct] = wmma16(pa, vbf, O[ct]);
    }
    __syncthreads();   // protect Qs/VTs before next stage
  }

  // ---- finalize: out = gamma * (O / l) + other ----
  const float g = gamma[0];
  float rl[8];
#pragma unroll
  for (int v = 0; v < 8; ++v) rl[v] = __frcp_rn(lrow[v]);
#pragma unroll
  for (int ct = 0; ct < 8; ++ct) {
#pragma unroll
    for (int v = 0; v < 8; ++v) {
      const int row = i0 + v + (half << 3);
      const int cch = (ct << 4) + l15;
      const float val = O[ct][v] * rl[v];
      const float res = (float)other[((size_t)b * N + row) * C + cch];
      out[((size_t)b * N + row) * C + cch] = (_Float16)(g * val + res);
    }
  }
}

// ===========================================================================
// Host orchestration
// ===========================================================================
extern "C" void kernel_launch(void* const* d_in, const int* in_sizes, int n_in,
                              void* d_out, int out_size, void* d_ws, size_t ws_size,
                              hipStream_t stream) {
  (void)in_sizes; (void)n_in; (void)out_size;

  // Flat input indices: setup_inputs() dict (file) order, nested dicts in
  // insertion order, lists in order.  _bb dicts: w, scale, bias, mean, var.
  enum {
    IDX_RGB = 0, IDX_DEPTH = 1,
    RD_DOWN_W = 2, RD_DOWN_B = 3, RD_BLK0 = 4, RD_BLK1 = 9, RD_BLK2 = 14,
    RD_BLK3 = 19, RD_FUSE = 24,
    ID_DOWN_W = 29, ID_DOWN_B = 30, ID_BLK0 = 31, ID_BLK1 = 36, ID_BLK2 = 41,
    ID_BLK3 = 46, ID_FUSE = 51,
    G_Q = 56, G_RGBK = 61, G_RGBV = 66, G_INFK = 71, G_INFV = 76,
    G_SECOND = 81, G_GAMMA1 = 86, G_GAMMA2 = 87
  };
  auto fptr = [&](int i) { return (const float*)d_in[i]; };

  // ---- workspace bump allocation (f16 elements) ----
  char* wsb = (char*)d_ws;
  size_t off = 0;
  auto alloc = [&](size_t elems) -> _Float16* {
    _Float16* p = (_Float16*)(wsb + off);
    off += elems * sizeof(_Float16);
    return p;
  };
  const size_t PIXC128 = (size_t)NPIX * 128;
  _Float16* xr    = alloc((size_t)NPIX * 256);   // rgb   NHWC f16
  _Float16* xd    = alloc((size_t)NPIX * 256);   // depth NHWC f16
  _Float16* downR = alloc((size_t)NPIX * 32);
  _Float16* FR    = alloc(PIXC128);              // dense features [o0|o1|o2|o3]
  _Float16* sumR  = alloc(PIXC128);
  _Float16* downI = alloc((size_t)NPIX * 32);
  _Float16* FI    = alloc(PIXC128);
  _Float16* sumI  = alloc(PIXC128);
  _Float16* q     = alloc(PIXC128);
  _Float16* kr    = alloc(PIXC128);
  _Float16* vr    = alloc(PIXC128);
  _Float16* ki    = alloc(PIXC128);
  _Float16* vi    = alloc(PIXC128);
  _Float16* refR  = alloc(PIXC128);
  _Float16* refI  = alloc(PIXC128);
  _Float16* wpk   = alloc(128 * 288 * 9);        // largest packed weight (fuse)
  if (off > ws_size) return;                     // not enough scratch: no-op

  // ---- activation layout conversion ----
  {
    const int tot = BATCH * 256 * PIX_PER_IMG;
    to_nhwc256<<<(tot + 255) / 256, 256, 0, stream>>>(fptr(IDX_RGB), xr);
    to_nhwc256<<<(tot + 255) / 256, 256, 0, stream>>>(fptr(IDX_DEPTH), xd);
  }

  // ---- conv launch helpers (pack then GEMM; stream-ordered) ----
  // NBLK: 2 for Cout=32 (wave covers all 32 channels), 4 for Cout=128.
  auto launch_conv = [&](const _Float16* iA, int cA, int sA, int oA,
                         const _Float16* iB, int cB, int sB, int oB,
                         int KS, int Cout,
                         const float* q0, const float* q1, const float* q2,
                         const float* q3, int mode,
                         _Float16* oH, int oS, int oO, float* oF) {
    if (Cout == 32)
      conv_gemm<2, 0><<<128, 256, 0, stream>>>(iA, cA, sA, oA, iB ? iB : iA, cB,
                                               sB, oB, wpk, KS, Cout, q0, q1, q2,
                                               q3, mode, oH, oS, oO, oF);
    else
      conv_gemm<4, 1><<<256, 256, 0, stream>>>(iA, cA, sA, oA, iB ? iB : iA, cB,
                                               sB, oB, wpk, KS, Cout, q0, q1, q2,
                                               q3, mode, oH, oS, oO, oF);
  };
  auto conv_bn = [&](int base, const _Float16* iA, int cA, int sA, int oA,
                     const _Float16* iB, int cB, int sB, int oB,
                     int KS, int Cout, _Float16* oH, int oS, int oO, float* oF) {
    const int Ci = cA + cB;
    const int tot = Cout * Ci * KS * KS;
    pack_weights<<<(tot + 255) / 256, 256, 0, stream>>>(fptr(base + 0), wpk, Cout, Ci, KS);
    launch_conv(iA, cA, sA, oA, iB, cB, sB, oB, KS, Cout,
                fptr(base + 1), fptr(base + 2), fptr(base + 3), fptr(base + 4),
                /*mode=*/1, oH, oS, oO, oF);
  };
  auto conv_bias = [&](int wIdx, int bIdx, const _Float16* iA, int cA, int sA,
                       int KS, int Cout, _Float16* oH, int oS, int oO) {
    const int tot = Cout * cA * KS * KS;
    pack_weights<<<(tot + 255) / 256, 256, 0, stream>>>(fptr(wIdx), wpk, Cout, cA, KS);
    launch_conv(iA, cA, sA, 0, nullptr, 0, sA, 0, KS, Cout,
                fptr(bIdx), fptr(bIdx), fptr(bIdx), fptr(bIdx),
                /*mode=*/0, oH, oS, oO, nullptr);
  };

  // ---- dense branch: down(1x1)+bias, 4 growing 3x3 BN-ReLU blocks, fuse ----
  auto run_dense = [&](int dw, int db, int b0, int b1, int b2, int b3, int fuse,
                       const _Float16* x, _Float16* down, _Float16* F,
                       _Float16* sum) {
    conv_bias(dw, db, x, 256, 256, 1, 32, down, 32, 0);
    conv_bn(b0, down, 32, 32, 0, nullptr, 0, 0, 0, 3, 32, F, 128, 0,  nullptr);
    conv_bn(b1, F, 32, 128, 0,  down, 32, 32, 0,  3, 32, F, 128, 32, nullptr);
    conv_bn(b2, F, 64, 128, 0,  down, 32, 32, 0,  3, 32, F, 128, 64, nullptr);
    conv_bn(b3, F, 96, 128, 0,  down, 32, 32, 0,  3, 32, F, 128, 96, nullptr);
    conv_bn(fuse, x, 256, 256, 0, F, 32, 128, 96, 3, 128, sum, 128, 0, nullptr);
  };
  run_dense(RD_DOWN_W, RD_DOWN_B, RD_BLK0, RD_BLK1, RD_BLK2, RD_BLK3, RD_FUSE,
            xr, downR, FR, sumR);
  run_dense(ID_DOWN_W, ID_DOWN_B, ID_BLK0, ID_BLK1, ID_BLK2, ID_BLK3, ID_FUSE,
            xd, downI, FI, sumI);

  // ---- GEFM ----
  conv_bn(G_Q,    sumR, 128, 128, 0, sumI, 128, 128, 0, 3, 128, q,  128, 0, nullptr);
  conv_bn(G_RGBK, sumR, 128, 128, 0, nullptr, 0, 0, 0,  3, 128, kr, 128, 0, nullptr);
  conv_bn(G_RGBV, sumR, 128, 128, 0, nullptr, 0, 0, 0,  3, 128, vr, 128, 0, nullptr);
  conv_bn(G_INFK, sumI, 128, 128, 0, nullptr, 0, 0, 0,  3, 128, ki, 128, 0, nullptr);
  conv_bn(G_INFV, sumI, 128, 128, 0, nullptr, 0, 0, 0,  3, 128, vi, 128, 0, nullptr);

  attend_flash<<<128, 256, 0, stream>>>(kr, q, vr, sumI, fptr(G_GAMMA1), refR);
  attend_flash<<<128, 256, 0, stream>>>(ki, q, vi, sumR, fptr(G_GAMMA2), refI);

  // final conv writes f32 NCHW straight into d_out
  conv_bn(G_SECOND, refR, 128, 128, 0, refI, 128, 128, 0, 3, 128,
          refR, 128, 0, (float*)d_out);
}
